// TGATEdgeClassifier_89910845375005
// MI455X (gfx1250) — matrix operand order repaired and checked by reading, not compiled, not run on hardware
//
#include <hip/hip_runtime.h>
#include <hip/hip_bf16.h>

typedef _Float16 f16;
typedef __attribute__((ext_vector_type(16))) _Float16 v16h;
typedef __attribute__((ext_vector_type(8)))  _Float16 v8h;
typedef __attribute__((ext_vector_type(2)))  _Float16 h2;
typedef __attribute__((ext_vector_type(8)))  float    v8f;
typedef __attribute__((ext_vector_type(4)))  float    v4f;

// concat two 16B half-chunks into one A fragment register set
static __device__ inline v16h cat8(v8h a, v8h b) {
  return __builtin_shufflevector(a, b, 0, 1, 2, 3, 4, 5, 6, 7,
                                        8, 9, 10, 11, 12, 13, 14, 15);
}
// convert 16 fp32 (two 32B chunks) into one f16 A fragment
static __device__ inline v16h cvtcat(v4f a, v4f b, v4f c, v4f d) {
  v16h r;
#pragma unroll
  for (int i = 0; i < 4; ++i) {
    r[i]      = (f16)a[i];
    r[4 + i]  = (f16)b[i];
    r[8 + i]  = (f16)c[i];
    r[12 + i] = (f16)d[i];
  }
  return r;
}

// ---------------------------------------------------------------------------
// Prep kernel (1 block): fold We/be into Wm1's edge block and pre-swizzle all
// B matrices into the WMMA B-fragment layout so the hot kernels fetch each
// lane's 16 halfs as one contiguous 32B LDS read.
//   fragment index i = ((frag*32 + lane)*16 + el)
//   frag = kt*4 + nt ;  k = kt*32 + (el>>3)*16 + (lane>>4)*8 + (el&7) ; n = nt*16 + (lane&15)
// ---------------------------------------------------------------------------
__global__ __launch_bounds__(256) void tgat_prep_kernel(
    const float* __restrict__ W1, const float* __restrict__ W2,
    const float* __restrict__ We, const float* __restrict__ be,
    const float* __restrict__ Wm1, const float* __restrict__ bm1,
    f16* __restrict__ w1f, f16* __restrict__ w2f,
    f16* __restrict__ wcatf, float* __restrict__ biasf)
{
  int tid = threadIdx.x;
  for (int i = tid; i < 4096; i += 256) {
    int el = i & 15, lane = (i >> 4) & 31, f = i >> 9;
    int kt = f >> 2, nt = f & 3;
    int k = kt * 32 + (el >> 3) * 16 + (lane >> 4) * 8 + (el & 7);
    int n = nt * 16 + (lane & 15);
    w1f[i] = (f16)W1[k * 64 + n];
    w2f[i] = (f16)W2[k * 64 + n];
  }
  // Folded head weight, K = 192 rows:
  //   k<128: Wm1 rows 0..127 (h_src|h_dst); k<160: We@Wm1[128:192]; else time rows
  for (int i = tid; i < 12288; i += 256) {
    int el = i & 15, lane = (i >> 4) & 31, f = i >> 9;
    int kt = f >> 2, nt = f & 3;
    int k = kt * 32 + (el >> 3) * 16 + (lane >> 4) * 8 + (el & 7);
    int n = nt * 16 + (lane & 15);
    float v;
    if (k < 128) {
      v = Wm1[k * 64 + n];
    } else if (k < 160) {
      int r = k - 128;
      v = 0.f;
      for (int j = 0; j < 64; ++j) v += We[r * 64 + j] * Wm1[(128 + j) * 64 + n];
    } else {
      v = Wm1[(192 + (k - 160)) * 64 + n];
    }
    wcatf[i] = (f16)v;
  }
  if (tid < 64) {
    float v = bm1[tid];
    for (int j = 0; j < 64; ++j) v += be[j] * Wm1[(128 + j) * 64 + tid];
    biasf[tid] = v;
  }
}

// ---------------------------------------------------------------------------
// Node encoder: h_nodes = relu(relu(x@W1+b1)@W2+b2) stored f16 in workspace.
// One 16-row tile per wave, 8 waves/block, both layers via WMMA f16.
// x is streamed non-temporally (read once).
// ---------------------------------------------------------------------------
__global__ __launch_bounds__(256) void tgat_node_kernel(
    const float* __restrict__ x, const float* __restrict__ b1,
    const float* __restrict__ b2, const f16* __restrict__ w1f,
    const f16* __restrict__ w2f, f16* __restrict__ hout, int ntiles)
{
  __shared__ __align__(16) f16 sW1[4096];
  __shared__ __align__(16) f16 sW2[4096];
  __shared__ __align__(16) f16 sH[8][16 * 64];

  int tid = threadIdx.x;
  {
    const uint4* s1 = (const uint4*)w1f;
    const uint4* s2 = (const uint4*)w2f;
    uint4* d1 = (uint4*)sW1;
    uint4* d2 = (uint4*)sW2;
    for (int i = tid; i < 512; i += 256) { d1[i] = s1[i]; d2[i] = s2[i]; }
  }
  __syncthreads();

  int wave = tid >> 5, lane = tid & 31;
  int tile = blockIdx.x * 8 + wave;
  if (tile >= ntiles) return;

  int r0 = tile * 16;
  int mrow = lane & 15;
  int half8 = (lane >> 4) * 8;

  // ---- layer 1: A from global x (fp32 -> f16, non-temporal), K = 64
  const float* xr = x + (size_t)(r0 + mrow) * 64;
  const v4f* xv0 = (const v4f*)(xr + half8);        // k: half8..half8+7
  const v4f* xv1 = (const v4f*)(xr + 16 + half8);   // k: 16+half8..+7
  const v4f* xv2 = (const v4f*)(xr + 32 + half8);
  const v4f* xv3 = (const v4f*)(xr + 48 + half8);
  v16h a0 = cvtcat(__builtin_nontemporal_load(xv0), __builtin_nontemporal_load(xv0 + 1),
                   __builtin_nontemporal_load(xv1), __builtin_nontemporal_load(xv1 + 1));
  v16h a1 = cvtcat(__builtin_nontemporal_load(xv2), __builtin_nontemporal_load(xv2 + 1),
                   __builtin_nontemporal_load(xv3), __builtin_nontemporal_load(xv3 + 1));

  v8f acc[4];
#pragma unroll
  for (int nt = 0; nt < 4; ++nt) {
    v8f c = {};
    v16h bA = *(const v16h*)&sW1[((0 * 4 + nt) * 32 + lane) * 16];
    c = __builtin_amdgcn_wmma_f32_16x16x32_f16(false, a0, false, bA, (short)0, c, false, false);
    v16h bB = *(const v16h*)&sW1[((1 * 4 + nt) * 32 + lane) * 16];
    c = __builtin_amdgcn_wmma_f32_16x16x32_f16(false, a1, false, bB, (short)0, c, false, false);
    acc[nt] = c;
  }
  f16* ht = sH[wave];
#pragma unroll
  for (int nt = 0; nt < 4; ++nt) {
    int n = nt * 16 + mrow;
    float bb = b1[n];
#pragma unroll
    for (int j = 0; j < 8; ++j) {
      float v = acc[nt][j] + bb;
      ht[(j + half8) * 64 + n] = (f16)(v > 0.f ? v : 0.f);
    }
  }
  __asm volatile("" ::: "memory");

  // ---- layer 2: A from LDS h tile (contiguous 16B chunks)
  v16h g0 = cat8(*(const v8h*)&ht[mrow * 64 + half8],
                 *(const v8h*)&ht[mrow * 64 + 16 + half8]);
  v16h g1 = cat8(*(const v8h*)&ht[mrow * 64 + 32 + half8],
                 *(const v8h*)&ht[mrow * 64 + 48 + half8]);
#pragma unroll
  for (int nt = 0; nt < 4; ++nt) {
    v8f c = {};
    v16h bA = *(const v16h*)&sW2[((0 * 4 + nt) * 32 + lane) * 16];
    c = __builtin_amdgcn_wmma_f32_16x16x32_f16(false, g0, false, bA, (short)0, c, false, false);
    v16h bB = *(const v16h*)&sW2[((1 * 4 + nt) * 32 + lane) * 16];
    c = __builtin_amdgcn_wmma_f32_16x16x32_f16(false, g1, false, bB, (short)0, c, false, false);
    acc[nt] = c;
  }
#pragma unroll
  for (int nt = 0; nt < 4; ++nt) {
    int n = nt * 16 + mrow;
    float bb = b2[n];
#pragma unroll
    for (int j = 0; j < 8; ++j) {
      float v = acc[nt][j] + bb;
      ht[(j + half8) * 64 + n] = (f16)(v > 0.f ? v : 0.f);
    }
  }
  __asm volatile("" ::: "memory");

  // coalesced store (keep RT caching: edge pass gathers from this table)
  {
    const uint4* s = (const uint4*)ht;  // 128 x uint4
    uint4* g = (uint4*)(hout + (size_t)r0 * 64);
#pragma unroll
    for (int q = 0; q < 4; ++q) g[lane * 4 + q] = s[lane * 4 + q];
  }
}

// ---------------------------------------------------------------------------
// Edge head: 32 edges per wave (2 M-subtiles share every B fragment):
//   A = [h_src | h_dst | edge_attr | time_enc] (f16, K=192 folded)
//   48 WMMAs/tile; head dot done in registers + __shfl_xor tree (no LDS z).
//   fp32 streams are non-temporal so h_nodes stays resident in L2.
// ---------------------------------------------------------------------------
__global__ __launch_bounds__(256) void tgat_edge_kernel(
    const int* __restrict__ srcN, const int* __restrict__ dstN,
    const float* __restrict__ eattr, const float* __restrict__ tenc,
    const f16* __restrict__ hnodes, const f16* __restrict__ wcatf,
    const float* __restrict__ biasf, const float* __restrict__ wm2,
    const float* __restrict__ bm2, float* __restrict__ out, int ntiles)
{
  __shared__ __align__(16) f16 sW[12288];  // 24 KB folded B fragments
  __shared__ float sWm2[64];
  __shared__ float sBias[64];

  int tid = threadIdx.x;
  {
    const uint4* s = (const uint4*)wcatf;
    uint4* d = (uint4*)sW;
    for (int i = tid; i < 1536; i += 256) d[i] = s[i];
    if (tid < 64) { sWm2[tid] = wm2[tid]; sBias[tid] = biasf[tid]; }
  }
  __syncthreads();

  int wave = tid >> 5, lane = tid & 31;
  int tile = blockIdx.x * 8 + wave;
  if (tile >= ntiles) return;

  int e0 = tile * 32;
  int mrow = lane & 15;
  int half8 = (lane >> 4) * 8;
  float bm2v = bm2[0];

  v16h A[2][6];
#pragma unroll
  for (int s = 0; s < 2; ++s) {
    int e = e0 + s * 16 + mrow;
    int si = srcN[e], di = dstN[e];
    const f16* hs = hnodes + (size_t)si * 64;
    const f16* hd = hnodes + (size_t)di * 64;
    const float* ea = eattr + (size_t)e * 32;
    const float* te = tenc + (size_t)e * 32;

    A[s][0] = cat8(*(const v8h*)(hs + half8), *(const v8h*)(hs + 16 + half8));
    A[s][1] = cat8(*(const v8h*)(hs + 32 + half8), *(const v8h*)(hs + 48 + half8));
    A[s][2] = cat8(*(const v8h*)(hd + half8), *(const v8h*)(hd + 16 + half8));
    A[s][3] = cat8(*(const v8h*)(hd + 32 + half8), *(const v8h*)(hd + 48 + half8));

    const v4f* ev0 = (const v4f*)(ea + half8);
    const v4f* ev1 = (const v4f*)(ea + 16 + half8);
    const v4f* tv0 = (const v4f*)(te + half8);
    const v4f* tv1 = (const v4f*)(te + 16 + half8);
    A[s][4] = cvtcat(__builtin_nontemporal_load(ev0), __builtin_nontemporal_load(ev0 + 1),
                     __builtin_nontemporal_load(ev1), __builtin_nontemporal_load(ev1 + 1));
    A[s][5] = cvtcat(__builtin_nontemporal_load(tv0), __builtin_nontemporal_load(tv0 + 1),
                     __builtin_nontemporal_load(tv1), __builtin_nontemporal_load(tv1 + 1));
  }

  float part0[8] = {0.f, 0.f, 0.f, 0.f, 0.f, 0.f, 0.f, 0.f};
  float part1[8] = {0.f, 0.f, 0.f, 0.f, 0.f, 0.f, 0.f, 0.f};

#pragma unroll
  for (int nt = 0; nt < 4; ++nt) {
    int n = nt * 16 + mrow;
    float bb = sBias[n];
    float ww = sWm2[n];
    v8f c0 = {}, c1 = {};
#pragma unroll
    for (int kt = 0; kt < 6; ++kt) {
      v16h b = *(const v16h*)&sW[((kt * 4 + nt) * 32 + lane) * 16];
      c0 = __builtin_amdgcn_wmma_f32_16x16x32_f16(false, A[0][kt], false, b, (short)0, c0, false, false);
      c1 = __builtin_amdgcn_wmma_f32_16x16x32_f16(false, A[1][kt], false, b, (short)0, c1, false, false);
    }
#pragma unroll
    for (int j = 0; j < 8; ++j) {
      float v0 = c0[j] + bb;
      float v1 = c1[j] + bb;
      part0[j] += (v0 > 0.f ? v0 : 0.f) * ww;
      part1[j] += (v1 > 0.f ? v1 : 0.f) * ww;
    }
  }

  // reduce over the 16-lane n-group (xor masks stay within the group)
#pragma unroll
  for (int j = 0; j < 8; ++j) {
    float p0 = part0[j], p1 = part1[j];
    p0 += __shfl_xor(p0, 1);  p1 += __shfl_xor(p1, 1);
    p0 += __shfl_xor(p0, 2);  p1 += __shfl_xor(p1, 2);
    p0 += __shfl_xor(p0, 4);  p1 += __shfl_xor(p1, 4);
    p0 += __shfl_xor(p0, 8);  p1 += __shfl_xor(p1, 8);
    part0[j] = p0; part1[j] = p1;
  }
  if ((lane & 15) == 0) {
    // this lane holds sums for rows m = half8 + j of each subtile
#pragma unroll
    for (int j = 0; j < 8; ++j) {
      __builtin_nontemporal_store(part0[j] + bm2v, out + e0 + half8 + j);
      __builtin_nontemporal_store(part1[j] + bm2v, out + e0 + 16 + half8 + j);
    }
  }
}

// ---------------------------------------------------------------------------
extern "C" void kernel_launch(void* const* d_in, const int* in_sizes, int n_in,
                              void* d_out, int out_size, void* d_ws, size_t ws_size,
                              hipStream_t stream) {
  const float* x    = (const float*)d_in[0];
  const int*   srcN = (const int*)d_in[1];
  const int*   dstN = (const int*)d_in[2];
  const float* ea   = (const float*)d_in[3];
  const float* te   = (const float*)d_in[4];
  const float* W1   = (const float*)d_in[5];
  const float* b1   = (const float*)d_in[6];
  const float* W2   = (const float*)d_in[7];
  const float* b2   = (const float*)d_in[8];
  const float* We   = (const float*)d_in[9];
  const float* be   = (const float*)d_in[10];
  const float* Wm1  = (const float*)d_in[11];
  const float* bm1  = (const float*)d_in[12];
  const float* Wm2  = (const float*)d_in[13];
  const float* bm2  = (const float*)d_in[14];
  float* out = (float*)d_out;

  const int N = in_sizes[0] / 64;  // FN = 64
  const int E = in_sizes[1];

  char* ws = (char*)d_ws;
  f16* hnodes = (f16*)ws;  // N*64 halfs
  size_t off = (size_t)N * 64 * sizeof(f16);
  off = (off + 255) & ~(size_t)255;
  f16* w1f   = (f16*)(ws + off); off += 4096 * sizeof(f16);
  f16* w2f   = (f16*)(ws + off); off += 4096 * sizeof(f16);
  f16* wcatf = (f16*)(ws + off); off += 12288 * sizeof(f16);
  float* biasf = (float*)(ws + off);

  tgat_prep_kernel<<<1, 256, 0, stream>>>(W1, W2, We, be, Wm1, bm1,
                                          w1f, w2f, wcatf, biasf);

  const int ntilesN = N / 16;  // 100000 / 16 = 6250
  const int blkN = (ntilesN + 7) / 8;
  tgat_node_kernel<<<blkN, 256, 0, stream>>>(x, b1, b2, w1f, w2f, hnodes, ntilesN);

  const int ntilesE = E / 32;  // 1e6 / 32 = 31250
  const int blkE = (ntilesE + 7) / 8;
  tgat_edge_kernel<<<blkE, 256, 0, stream>>>(srcN, dstN, ea, te, hnodes, wcatf,
                                             biasf, Wm2, bm2, out, ntilesE);
}
